// ManyChannelsIntegrator_53051436040338
// MI455X (gfx1250) — matrix operand order repaired and checked by reading, not compiled, not run on hardware
//
#include <hip/hip_runtime.h>

#define NN 2048
#define BB 256
#define TT 256
#define SLOPE 8.0f

typedef __attribute__((ext_vector_type(16))) __bf16 v16bf;
typedef __attribute__((ext_vector_type(8)))  float  v8f;

union Frag16 {
    uint4  u[2];
    v16bf  v;
};

__device__ __forceinline__ unsigned int f2bf(float f) {
    unsigned int u = __float_as_uint(f);
    u += 0x7FFFu + ((u >> 16) & 1u);   // round-to-nearest-even
    return u >> 16;
}

// ---------------- prep kernels (once per launch) ----------------

// Convert W (fp32 [N,N], row-major) to bf16 row-major. Row n of W is column n
// of the WMMA B-matrix; step_kernel loads each B fragment per-lane contiguous in
// K, so row-major bf16 W needs no further repacking.
__global__ void prep_w_bf16(const float* __restrict__ W,
                            unsigned short* __restrict__ Wb) {
    int i = blockIdx.x * blockDim.x + threadIdx.x;   // over N*N/4
    const float4 w = ((const float4*)W)[i];
    uint2 o;
    o.x = f2bf(w.x) | (f2bf(w.y) << 16);
    o.y = f2bf(w.z) | (f2bf(w.w) << 16);
    ((uint2*)Wb)[i] = o;
}

// me[c][n] = mask[n] * encoders[c][n]
__global__ void prep_me(const float* __restrict__ mask,
                        const float* __restrict__ enc,
                        float* __restrict__ me) {
    int i = blockIdx.x * blockDim.x + threadIdx.x;   // over 2*N
    me[i] = mask[i & (NN - 1)] * enc[i];
}

__global__ void zero_out(float* __restrict__ o) {
    int i = blockIdx.x * blockDim.x + threadIdx.x;
    o[i] = 0.0f;
}

// ---------------- per-timestep GEMM + activation + decode ----------------
// grid (8, 16): blockIdx.x -> 32-row m-chunk, blockIdx.y -> 128-col n-chunk.
// 256 threads = 8 waves; wave w: m-tile = (w&1), 2 n-tiles at (w>>1)*32.
__global__ __launch_bounds__(256) void step_kernel(
    const float* __restrict__ state_prev,   // [B,N]
    float* __restrict__ state_next,         // [B,N]
    const unsigned short* __restrict__ Wb,  // [N,N] bf16, row-major (L2 resident)
    const float* __restrict__ me,           // [2,N] mask*encoders
    const float* __restrict__ inputs,       // [2,B,T]
    const float* __restrict__ bias,
    const float* __restrict__ thr,
    const float* __restrict__ mask,
    const float* __restrict__ dec,          // [2,N]
    float* __restrict__ outs,               // [2,B,T]
    int t)
{
    extern __shared__ unsigned short uPan[];  // 32 x 2048 bf16 = 128 KB
    const int tid   = threadIdx.x;
    const int mbase = blockIdx.x * 32;
    const int nbase = blockIdx.y * 128;

    // ---- stage A panel: u[b,k] = state[b,k] + x0[b]*me0[k] + x1[b]*me1[k] ----
    {
        const int row = tid >> 3;            // 0..31
        const int cb  = (tid & 7) * 256;     // 256 contiguous columns
        const int b   = mbase + row;
        const float x0 = inputs[b * TT + t];
        const float x1 = inputs[BB * TT + b * TT + t];
        const float4* sp4 = (const float4*)(state_prev + (size_t)b * NN + cb);
        const float4* m04 = (const float4*)(me + cb);
        const float4* m14 = (const float4*)(me + NN + cb);
        uint2* dst = (uint2*)(uPan + row * NN + cb);
        #pragma unroll 4
        for (int c4 = 0; c4 < 64; ++c4) {
            const float4 s  = sp4[c4];
            const float4 e0 = m04[c4];
            const float4 e1 = m14[c4];
            float4 u;
            u.x = s.x + x0 * e0.x + x1 * e1.x;
            u.y = s.y + x0 * e0.y + x1 * e1.y;
            u.z = s.z + x0 * e0.z + x1 * e1.z;
            u.w = s.w + x0 * e0.w + x1 * e1.w;
            uint2 o;
            o.x = f2bf(u.x) | (f2bf(u.y) << 16);
            o.y = f2bf(u.z) | (f2bf(u.w) << 16);
            dst[c4] = o;
        }
    }
    __syncthreads();

    const int w    = tid >> 5;
    const int lane = tid & 31;
    const int hi   = lane >> 4;       // half-wave select
    const int lx   = lane & 15;
    const int mrow  = (w & 1) * 16;           // wave's row-tile inside panel
    const int ncol0 = nbase + (w >> 1) * 32;  // wave's first n column (2 tiles)

    const v8f vzero = {0.f, 0.f, 0.f, 0.f, 0.f, 0.f, 0.f, 0.f};
    v8f acc0 = vzero;
    v8f acc1 = vzero;

    // A fragment: lane holds M=lx, K in
    // [kk + hi*16 .. +16), contiguous (matches B).
    const unsigned short* ap = uPan + (mrow + lx) * NN + hi * 16;
    // B fragment (contiguous K, same order as A): lane holds N=lx, K in
    // [kk + hi*16 .. +16), contiguous = row-major W row.
    const unsigned short* p0 = Wb + (size_t)(ncol0 + lx) * NN + hi * 16;
    const unsigned short* p1 = Wb + (size_t)(ncol0 + 16 + lx) * NN + hi * 16;

    #pragma unroll 1
    for (int kk = 0; kk < NN; kk += 64) {
        Frag16 a0, a1, b00, b01, b10, b11;
        a0.u[0]  = *(const uint4*)(ap);          // ds_load_b128
        a0.u[1]  = *(const uint4*)(ap + 8);
        a1.u[0]  = *(const uint4*)(ap + 32);
        a1.u[1]  = *(const uint4*)(ap + 40);
        b00.u[0] = *(const uint4*)(p0);          // global_load_b128 (L2 hit)
        b00.u[1] = *(const uint4*)(p0 + 8);
        b10.u[0] = *(const uint4*)(p1);
        b10.u[1] = *(const uint4*)(p1 + 8);
        b01.u[0] = *(const uint4*)(p0 + 32);
        b01.u[1] = *(const uint4*)(p0 + 40);
        b11.u[0] = *(const uint4*)(p1 + 32);
        b11.u[1] = *(const uint4*)(p1 + 40);
        // Speculative prefetch 256B/lane ahead (pull toward near caches);
        // safe past the end of a row (ws is contiguous, dropped on fault).
        __builtin_prefetch(p0 + 128, 0, 3);
        __builtin_prefetch(p1 + 128, 0, 3);
        acc0 = __builtin_amdgcn_wmma_f32_16x16x32_bf16(
            false, a0.v, false, b00.v, (short)0, acc0, false, false);
        acc1 = __builtin_amdgcn_wmma_f32_16x16x32_bf16(
            false, a0.v, false, b10.v, (short)0, acc1, false, false);
        acc0 = __builtin_amdgcn_wmma_f32_16x16x32_bf16(
            false, a1.v, false, b01.v, (short)0, acc0, false, false);
        acc1 = __builtin_amdgcn_wmma_f32_16x16x32_bf16(
            false, a1.v, false, b11.v, (short)0, acc1, false, false);
        ap += 64;
        p0 += 64;
        p1 += 64;
    }

    // ---- epilogue: sigmoid, state write, fused decode ----
    // C/D layout: element (m = j + 8*hi, n = lx) in acc vgpr j.
    const int brow0 = mbase + mrow + hi * 8;
    #pragma unroll
    for (int q = 0; q < 2; ++q) {
        const v8f acc = q ? acc1 : acc0;
        const int n  = ncol0 + q * 16 + lx;
        const float mk = mask[n];
        const float bs = bias[n];
        const float th = thr[n];
        const float d0 = dec[n];
        const float d1 = dec[NN + n];
        #pragma unroll
        for (int j = 0; j < 8; ++j) {
            const int b = brow0 + j;
            const float pre = acc[j];
            // sigmoid via v_exp_f32 + v_rcp_f32 (avoid the IEEE divide
            // expansion: div_scale/rcp/4xFMA/div_fmas per element).
            const float e  = __expf(-SLOPE * (pre - th));
            const float sg = __builtin_amdgcn_rcpf(1.0f + e);
            const float s  = mk * (bs + sg);
            state_next[(size_t)b * NN + n] = s;
            // decode: outs[c,b,t] += sum_n s*dec[c,n]; reduce over the 16
            // lanes of this half-wave (each holds a distinct n).
            float p0v = s * d0;
            float p1v = s * d1;
            #pragma unroll
            for (int off = 8; off > 0; off >>= 1) {
                p0v += __shfl_xor(p0v, off, 16);
                p1v += __shfl_xor(p1v, off, 16);
            }
            if (lx == 0) {
                atomicAdd(outs + b * TT + t, p0v);
                atomicAdd(outs + BB * TT + b * TT + t, p1v);
            }
        }
    }
}

// ---------------- host launcher ----------------
extern "C" void kernel_launch(void* const* d_in, const int* in_sizes, int n_in,
                              void* d_out, int out_size, void* d_ws, size_t ws_size,
                              hipStream_t stream) {
    const float* inputs = (const float*)d_in[0];   // [2,256,256]
    const float* state0 = (const float*)d_in[1];   // [256,2048]
    const float* mask   = (const float*)d_in[2];   // [2048]
    const float* enc    = (const float*)d_in[3];   // [2,2048]
    const float* dec    = (const float*)d_in[4];   // [2,2048]
    const float* W      = (const float*)d_in[5];   // [2048,2048]
    const float* bias   = (const float*)d_in[6];   // [2048]
    const float* thr    = (const float*)d_in[7];   // [2048]
    float* outs = (float*)d_out;                   // [2,256,256]

    char* ws = (char*)d_ws;
    unsigned short* Wb = (unsigned short*)ws;                         // 8 MB
    float* me  = (float*)(ws + (size_t)NN * NN * 2);                  // 16 KB
    float* stA = (float*)(ws + (size_t)NN * NN * 2 + 2 * NN * 4);     // 2 MB
    float* stB = stA + (size_t)BB * NN;                               // 2 MB

    prep_w_bf16<<<(NN * NN / 4) / 256, 256, 0, stream>>>(W, Wb);
    prep_me<<<(2 * NN) / 256, 256, 0, stream>>>(mask, enc, me);
    zero_out<<<(2 * BB * TT) / 256, 256, 0, stream>>>(outs);

    const size_t shmem = 32 * NN * sizeof(unsigned short);  // 128 KB dynamic LDS
    const float* prev = state0;
    float* bufs[2] = {stA, stB};
    for (int t = 0; t < TT; ++t) {
        float* next = bufs[t & 1];
        step_kernel<<<dim3(8, 16), 256, shmem, stream>>>(
            prev, next, Wb, me, inputs, bias, thr, mask, dec, outs, t);
        prev = next;
    }
}
